// IPMP_37134287242033
// MI455X (gfx1250) — compile-verified, matrix operand order checked by reading, not run on hardware
//
#include <hip/hip_runtime.h>
#include <math.h>

#define N_NODES 20000
#define N_EDGES 640000
#define CS 128
#define CZ 64
#define CH 128
#define PP 8
#define PRE 448
#define EPSC 1e-8f
#define LN_EPS 1e-5f

typedef __attribute__((ext_vector_type(16))) _Float16 v16h;
typedef __attribute__((ext_vector_type(8)))  _Float16 v8h;
typedef __attribute__((ext_vector_type(4)))  _Float16 v4h;
typedef __attribute__((ext_vector_type(8)))  float    v8f;

__device__ __forceinline__ float wred(float v) {
    #pragma unroll
    for (int o = 16; o > 0; o >>= 1) v += __shfl_xor(v, o, 32);
    return v;
}

__device__ __forceinline__ v4h pack4(float x, float y, float z, float w) {
    v4h p;
    p[0] = (_Float16)x; p[1] = (_Float16)y;
    p[2] = (_Float16)z; p[3] = (_Float16)w;
    return p;
}

// A fragment: two contiguous 8xf16 runs from a row (LDS) -> v16h
__device__ __forceinline__ v16h load_afrag(const _Float16* row, int ks, int kh) {
    const _Float16* p = row + ks * 32 + kh * 8;
    v8h lo = *(const v8h*)(p);
    v8h hi = *(const v8h*)(p + 16);
    return __builtin_shufflevector(lo, hi, 0, 1, 2, 3, 4, 5, 6, 7,
                                   8, 9, 10, 11, 12, 13, 14, 15);
}

// B fragment: pre-packed, one contiguous 16xf16 (32B) vector per lane
__device__ __forceinline__ v16h load_bfrag(const _Float16* wp, long tile, int lane) {
    return *(const v16h*)(wp + (tile * 32 + lane) * 16);
}

// ---------------- pack f32 weight [K][N] into WMMA B-fragment order (f16) ----
__global__ void pack_b_kernel(const float* __restrict__ src,
                              _Float16* __restrict__ dst, int K, int N) {
    int idx = blockIdx.x * 256 + threadIdx.x;
    if (idx >= K * N) return;
    int i    = idx & 15;
    int lane = (idx >> 4) & 31;
    int t    = idx >> 9;
    int ntiles = N >> 4;
    int nt = t % ntiles;
    int ks = t / ntiles;
    int kh = lane >> 4, lm = lane & 15;
    int v = i >> 1, odd = i & 1;
    int kk  = ks * 32 + kh * 8 + ((v < 4) ? 2 * v : 16 + 2 * (v - 4)) + odd;
    int col = nt * 16 + lm;
    dst[idx] = (_Float16)src[kk * N + col];
}

// ---------------------------------------------------------------- edge count
__global__ void cnt_kernel(const int* __restrict__ ei, float* __restrict__ cnt) {
    long e = (long)blockIdx.x * 256 + threadIdx.x;
    if (e < N_EDGES) atomicAdd(&cnt[ei[N_EDGES + e]], 1.0f);
}

// ------------------------------------------------- per-node point projection
__global__ void node_pts_kernel(const float* __restrict__ s,
                                const float* __restrict__ w,
                                const float* __restrict__ b,
                                const float* __restrict__ rot,
                                const float* __restrict__ trans,
                                float* __restrict__ ptsl,
                                float* __restrict__ ptsg) {
    __shared__ float plbuf[8][24];
    int wave = threadIdx.x >> 5, lane = threadIdx.x & 31;
    long n = (long)blockIdx.x * 8 + wave;
    bool ok = (n < N_NODES);
    if (ok && lane < 24) {
        float acc = b[lane];
        for (int k = 0; k < CS; ++k) acc += s[n * CS + k] * w[k * 24 + lane];
        plbuf[wave][lane] = acc;
        ptsl[n * 24 + lane] = acc;
    }
    __syncthreads();
    if (ok && lane < 24) {
        int p = lane / 3, i = lane % 3;
        float pg = rot[n * 9 + i * 3 + 0] * plbuf[wave][p * 3 + 0]
                 + rot[n * 9 + i * 3 + 1] * plbuf[wave][p * 3 + 1]
                 + rot[n * 9 + i * 3 + 2] * plbuf[wave][p * 3 + 2]
                 + 0.1f * trans[n * 3 + i];
        ptsg[n * 24 + lane] = pg;
    }
}

// --------------------------------------- fused premessage + MLP3 (edge tile)
template <bool NODEPATH>
__global__ void edge_mlp_kernel(const float* __restrict__ s,
                                const float* __restrict__ zin,
                                const int* __restrict__ ei,
                                const float* __restrict__ rot,
                                const float* __restrict__ trans,
                                const float* __restrict__ ptsl,
                                const float* __restrict__ ptsg,
                                const _Float16* __restrict__ w1, const float* __restrict__ b1,
                                const _Float16* __restrict__ w2, const float* __restrict__ b2,
                                const _Float16* __restrict__ w3, const float* __restrict__ b3,
                                const float* __restrict__ lng, const float* __restrict__ lnb,
                                float* __restrict__ upd, float* __restrict__ zout) {
    constexpr int CO = NODEPATH ? CS : CZ;
    __shared__ __align__(16) _Float16 pre[16][PRE];
    __shared__ __align__(16) _Float16 h1[16][CH];
    __shared__ __align__(16) _Float16 h2[16][CH];
    __shared__ float pgs[16][PP][3];
    __shared__ float pgd[16][PP][3];
    __shared__ int   ssrc[16];
    __shared__ float msgf[16][CZ];

    int tid = threadIdx.x;
    {   // ---- build premessage: 16 threads per edge, float4-vectorized ----
        int m = tid >> 4, g = tid & 15;
        long e = (long)blockIdx.x * 16 + m;
        int sidx = ei[N_EDGES + e];   // src
        int didx = ei[e];             // dst
        if (g == 0) ssrc[m] = sidx;
        const float4* ssrc4 = (const float4*)(s + (long)sidx * CS);
        const float4* sdst4 = (const float4*)(s + (long)didx * CS);
        #pragma unroll
        for (int j4 = g; j4 < CS / 4; j4 += 16) {
            float4 av = ssrc4[j4];
            float4 bv = sdst4[j4];
            *(v4h*)&pre[m][j4 * 4]      = pack4(av.x, av.y, av.z, av.w);
            *(v4h*)&pre[m][CS + j4 * 4] = pack4(bv.x, bv.y, bv.z, bv.w);
        }
        {
            float4 zv = ((const float4*)(zin + e * CZ))[g];
            *(v4h*)&pre[m][2 * CS + g * 4] = pack4(zv.x, zv.y, zv.z, zv.w);
        }
        if (g < PP) {
            int p = g;
            float pl0 = ptsl[(long)sidx * 24 + p * 3 + 0];
            float pl1 = ptsl[(long)sidx * 24 + p * 3 + 1];
            float pl2 = ptsl[(long)sidx * 24 + p * 3 + 2];
            pre[m][320 + p * 3 + 0] = (_Float16)pl0;
            pre[m][320 + p * 3 + 1] = (_Float16)pl1;
            pre[m][320 + p * 3 + 2] = (_Float16)pl2;
            float a0 = pl0 + EPSC, a1 = pl1 + EPSC, a2 = pl2 + EPSC;
            pre[m][344 + p] = (_Float16)sqrtf(a0 * a0 + a1 * a1 + a2 * a2);
            float gs0 = ptsg[(long)sidx * 24 + p * 3 + 0];
            float gs1 = ptsg[(long)sidx * 24 + p * 3 + 1];
            float gs2 = ptsg[(long)sidx * 24 + p * 3 + 2];
            pgs[m][p][0] = gs0; pgs[m][p][1] = gs1; pgs[m][p][2] = gs2;
            float gd0 = ptsg[(long)didx * 24 + p * 3 + 0];
            float gd1 = ptsg[(long)didx * 24 + p * 3 + 1];
            float gd2 = ptsg[(long)didx * 24 + p * 3 + 2];
            pgd[m][p][0] = gd0; pgd[m][p][1] = gd1; pgd[m][p][2] = gd2;
            float v0 = gd0 - 0.1f * trans[sidx * 3 + 0];
            float v1 = gd1 - 0.1f * trans[sidx * 3 + 1];
            float v2 = gd2 - 0.1f * trans[sidx * 3 + 2];
            const float* R = rot + (long)sidx * 9;
            float d0 = R[0] * v0 + R[3] * v1 + R[6] * v2;
            float d1 = R[1] * v0 + R[4] * v1 + R[7] * v2;
            float d2 = R[2] * v0 + R[5] * v1 + R[8] * v2;
            pre[m][352 + p * 3 + 0] = (_Float16)d0;
            pre[m][352 + p * 3 + 1] = (_Float16)d1;
            pre[m][352 + p * 3 + 2] = (_Float16)d2;
            float e0 = d0 + EPSC, e1 = d1 + EPSC, e2 = d2 + EPSC;
            pre[m][376 + p] = (_Float16)sqrtf(e0 * e0 + e1 * e1 + e2 * e2);
        }
    }
    __syncthreads();
    for (int idx = tid; idx < 16 * 64; idx += 256) {
        int mm = idx >> 6, pq = idx & 63, p = pq >> 3, q = pq & 7;
        float dx = pgs[mm][p][0] - pgd[mm][q][0] + EPSC;
        float dy = pgs[mm][p][1] - pgd[mm][q][1] + EPSC;
        float dz = pgs[mm][p][2] - pgd[mm][q][2] + EPSC;
        pre[mm][384 + pq] = (_Float16)sqrtf(dx * dx + dy * dy + dz * dz);
    }
    __syncthreads();

    int wave = tid >> 5, lane = tid & 31;
    int lm = lane & 15, kh = lane >> 4;

    v8f acc1 = {};
    for (int ks = 0; ks < PRE / 32; ++ks) {
        v16h a = load_afrag(&pre[lm][0], ks, kh);
        v16h b = load_bfrag(w1, (long)ks * (CH / 16) + wave, lane);
        acc1 = __builtin_amdgcn_wmma_f32_16x16x32_f16(false, a, false, b, (short)0, acc1, false, false);
    }
    {
        int n = wave * 16 + lm; float bb = b1[n];
        #pragma unroll
        for (int r = 0; r < 8; ++r) {
            int mm = r + 8 * kh;
            h1[mm][n] = (_Float16)fmaxf(acc1[r] + bb, 0.0f);
        }
    }
    __syncthreads();

    v8f acc2 = {};
    for (int ks = 0; ks < CH / 32; ++ks) {
        v16h a = load_afrag(&h1[lm][0], ks, kh);
        v16h b = load_bfrag(w2, (long)ks * (CH / 16) + wave, lane);
        acc2 = __builtin_amdgcn_wmma_f32_16x16x32_f16(false, a, false, b, (short)0, acc2, false, false);
    }
    {
        int n = wave * 16 + lm; float bb = b2[n];
        #pragma unroll
        for (int r = 0; r < 8; ++r) {
            int mm = r + 8 * kh;
            h2[mm][n] = (_Float16)fmaxf(acc2[r] + bb, 0.0f);
        }
    }
    __syncthreads();

    if (wave * 16 < CO) {
        v8f acc3 = {};
        for (int ks = 0; ks < CH / 32; ++ks) {
            v16h a = load_afrag(&h2[lm][0], ks, kh);
            v16h b = load_bfrag(w3, (long)ks * (CO / 16) + wave, lane);
            acc3 = __builtin_amdgcn_wmma_f32_16x16x32_f16(false, a, false, b, (short)0, acc3, false, false);
        }
        int n = wave * 16 + lm; float bb = b3[n];
        if constexpr (NODEPATH) {
            #pragma unroll
            for (int r = 0; r < 8; ++r) {
                int mm = r + 8 * kh;
                atomicAdd(&upd[(long)ssrc[mm] * CS + n], acc3[r] + bb);
            }
        } else {
            #pragma unroll
            for (int r = 0; r < 8; ++r) {
                int mm = r + 8 * kh;
                msgf[mm][n] = acc3[r] + bb;
            }
        }
    }
    if constexpr (!NODEPATH) {   // z = LN(z + msg)
        __syncthreads();
        #pragma unroll
        for (int rr = 0; rr < 2; ++rr) {
            int mm = wave * 2 + rr;
            long ee = (long)blockIdx.x * 16 + mm;
            float x1 = zin[ee * CZ + lane]      + msgf[mm][lane];
            float x2 = zin[ee * CZ + lane + 32] + msgf[mm][lane + 32];
            float sm = wred(x1 + x2);
            float sq = wred(x1 * x1 + x2 * x2);
            float mean = sm * (1.0f / 64.0f);
            float var  = sq * (1.0f / 64.0f) - mean * mean;
            float inv  = rsqrtf(var + LN_EPS);
            zout[ee * CZ + lane]      = (x1 - mean) * inv * lng[lane]      + lnb[lane];
            zout[ee * CZ + lane + 32] = (x2 - mean) * inv * lng[lane + 32] + lnb[lane + 32];
        }
    }
}

// -------------------------------------------- s = LN(s + upd/cnt * mask)
// one wave per node; each lane owns a contiguous float4 slice of 128 channels
__global__ void node_ln1_kernel(const float* __restrict__ s,
                                const float* __restrict__ upd,
                                const float* __restrict__ cnt,
                                const float* __restrict__ mask,
                                const float* __restrict__ g1,
                                const float* __restrict__ b1,
                                float* __restrict__ smid) {
    int wave = threadIdx.x >> 5, lane = threadIdx.x & 31;
    long n = (long)blockIdx.x * 8 + wave;
    if (n >= N_NODES) return;
    float c = fmaxf(cnt[n], 1.0f);
    float mk = mask[n];
    float4 sv = ((const float4*)(s + n * CS))[lane];
    float4 uv = ((const float4*)(upd + n * CS))[lane];
    float x0 = sv.x + (uv.x / c) * mk;
    float x1 = sv.y + (uv.y / c) * mk;
    float x2 = sv.z + (uv.z / c) * mk;
    float x3 = sv.w + (uv.w / c) * mk;
    float sm = wred(x0 + x1 + x2 + x3);
    float sq = wred(x0 * x0 + x1 * x1 + x2 * x2 + x3 * x3);
    float mean = sm * (1.0f / 128.0f);
    float var  = sq * (1.0f / 128.0f) - mean * mean;
    float inv  = rsqrtf(var + LN_EPS);
    float4 gv = ((const float4*)g1)[lane];
    float4 bv = ((const float4*)b1)[lane];
    float4 ov;
    ov.x = (x0 - mean) * inv * gv.x + bv.x;
    ov.y = (x1 - mean) * inv * gv.y + bv.y;
    ov.z = (x2 - mean) * inv * gv.z + bv.z;
    ov.w = (x3 - mean) * inv * gv.w + bv.w;
    ((float4*)(smid + n * CS))[lane] = ov;
}

// ------------------- FFN (128->512 GELU ->128) + residual + LN2 via WMMA
__global__ void ffn_kernel(const float* __restrict__ smid,
                           const _Float16* __restrict__ fw1, const float* __restrict__ fb1,
                           const _Float16* __restrict__ fw2, const float* __restrict__ fb2,
                           const float* __restrict__ g2, const float* __restrict__ b2ln,
                           float* __restrict__ sfinal) {
    __shared__ __align__(16) _Float16 a16[16][CS];
    __shared__ __align__(16) _Float16 hf[16][4 * CS];
    __shared__ __align__(16) float outf[16][CS];
    int tid = threadIdx.x;
    long nbase = (long)blockIdx.x * 16;
    for (int idx = tid; idx < 16 * CS / 4; idx += 256) {
        int mm = idx >> 5, c4 = idx & 31;
        float4 v = ((const float4*)(smid + (nbase + mm) * CS))[c4];
        *(v4h*)&a16[mm][c4 * 4] = pack4(v.x, v.y, v.z, v.w);
    }
    __syncthreads();
    int wave = tid >> 5, lane = tid & 31, lm = lane & 15, kh = lane >> 4;

    for (int nt = wave; nt < 32; nt += 8) {
        v8f acc = {};
        for (int ks = 0; ks < CS / 32; ++ks) {
            v16h a = load_afrag(&a16[lm][0], ks, kh);
            v16h b = load_bfrag(fw1, (long)ks * 32 + nt, lane);
            acc = __builtin_amdgcn_wmma_f32_16x16x32_f16(false, a, false, b, (short)0, acc, false, false);
        }
        int n = nt * 16 + lm; float bb = fb1[n];
        #pragma unroll
        for (int r = 0; r < 8; ++r) {
            int mm = r + 8 * kh;
            float xv = acc[r] + bb;
            hf[mm][n] = (_Float16)(0.5f * xv * (1.0f + erff(xv * 0.70710678f)));
        }
    }
    __syncthreads();
    {
        v8f acc = {};
        for (int ks = 0; ks < 512 / 32; ++ks) {
            v16h a = load_afrag(&hf[lm][0], ks, kh);
            v16h b = load_bfrag(fw2, (long)ks * (CS / 16) + wave, lane);
            acc = __builtin_amdgcn_wmma_f32_16x16x32_f16(false, a, false, b, (short)0, acc, false, false);
        }
        int n = wave * 16 + lm; float bb = fb2[n];
        #pragma unroll
        for (int r = 0; r < 8; ++r) {
            int mm = r + 8 * kh;
            outf[mm][n] = acc[r] + bb + smid[(nbase + mm) * CS + n];
        }
    }
    __syncthreads();
    #pragma unroll
    for (int rr = 0; rr < 2; ++rr) {
        int mm = wave * 2 + rr;
        long n = nbase + mm;
        float4 v = *(const float4*)&outf[mm][lane * 4];
        float sm = wred(v.x + v.y + v.z + v.w);
        float sq = wred(v.x * v.x + v.y * v.y + v.z * v.z + v.w * v.w);
        float mean = sm * (1.0f / 128.0f);
        float var  = sq * (1.0f / 128.0f) - mean * mean;
        float inv  = rsqrtf(var + LN_EPS);
        float4 gv = ((const float4*)g2)[lane];
        float4 bv = ((const float4*)b2ln)[lane];
        float4 ov;
        ov.x = (v.x - mean) * inv * gv.x + bv.x;
        ov.y = (v.y - mean) * inv * gv.y + bv.y;
        ov.z = (v.z - mean) * inv * gv.z + bv.z;
        ov.w = (v.w - mean) * inv * gv.w + bv.w;
        ((float4*)(sfinal + n * CS))[lane] = ov;
    }
}

// ============================================================== launch
extern "C" void kernel_launch(void* const* d_in, const int* in_sizes, int n_in,
                              void* d_out, int out_size, void* d_ws, size_t ws_size,
                              hipStream_t stream) {
    const float* s     = (const float*)d_in[0];
    const float* z     = (const float*)d_in[1];
    const int*   ei    = (const int*)d_in[2];
    const float* rot   = (const float*)d_in[3];
    const float* trans = (const float*)d_in[4];
    const float* mask  = (const float*)d_in[5];
    const float* node_pts_w = (const float*)d_in[6];
    const float* node_pts_b = (const float*)d_in[7];
    const float* nm_w1 = (const float*)d_in[8];
    const float* nm_b1 = (const float*)d_in[9];
    const float* nm_w2 = (const float*)d_in[10];
    const float* nm_b2 = (const float*)d_in[11];
    const float* nm_w3 = (const float*)d_in[12];
    const float* nm_b3 = (const float*)d_in[13];
    const float* ln1_g = (const float*)d_in[14];
    const float* ln1_b = (const float*)d_in[15];
    const float* ffn_w1 = (const float*)d_in[16];
    const float* ffn_b1 = (const float*)d_in[17];
    const float* ffn_w2 = (const float*)d_in[18];
    const float* ffn_b2 = (const float*)d_in[19];
    const float* ln2_g = (const float*)d_in[20];
    const float* ln2_b = (const float*)d_in[21];
    const float* edge_pts_w = (const float*)d_in[22];
    const float* edge_pts_b = (const float*)d_in[23];
    const float* em_w1 = (const float*)d_in[24];
    const float* em_b1 = (const float*)d_in[25];
    const float* em_w2 = (const float*)d_in[26];
    const float* em_b2 = (const float*)d_in[27];
    const float* em_w3 = (const float*)d_in[28];
    const float* em_b3 = (const float*)d_in[29];
    const float* lne_g = (const float*)d_in[30];
    const float* lne_b = (const float*)d_in[31];

    char* wsb = (char*)d_ws;
    size_t off = 0;
    auto alloc = [&](size_t bytes) {
        void* p = wsb + off;
        off = (off + bytes + 255) & ~(size_t)255;
        return p;
    };
    _Float16* w1n = (_Float16*)alloc((size_t)PRE * CH * 2);
    _Float16* w2n = (_Float16*)alloc((size_t)CH * CH * 2);
    _Float16* w3n = (_Float16*)alloc((size_t)CH * CS * 2);
    _Float16* w1e = (_Float16*)alloc((size_t)PRE * CH * 2);
    _Float16* w2e = (_Float16*)alloc((size_t)CH * CH * 2);
    _Float16* w3e = (_Float16*)alloc((size_t)CH * CZ * 2);
    _Float16* fw1 = (_Float16*)alloc((size_t)CS * 4 * CS * 2);
    _Float16* fw2 = (_Float16*)alloc((size_t)4 * CS * CS * 2);
    float* ptsl = (float*)alloc((size_t)N_NODES * 24 * 4);
    float* ptsg = (float*)alloc((size_t)N_NODES * 24 * 4);
    float* upd  = (float*)alloc((size_t)N_NODES * CS * 4);
    float* cnt  = (float*)alloc((size_t)N_NODES * 4);
    float* smid = (float*)alloc((size_t)N_NODES * CS * 4);

    float* out_s = (float*)d_out;
    float* out_z = out_s + (size_t)N_NODES * CS;

    hipMemsetAsync(upd, 0, (size_t)N_NODES * CS * 4, stream);
    hipMemsetAsync(cnt, 0, (size_t)N_NODES * 4, stream);

    auto pack = [&](const float* src, _Float16* dst, int K, int N) {
        pack_b_kernel<<<(K * N + 255) / 256, 256, 0, stream>>>(src, dst, K, N);
    };
    pack(nm_w1, w1n, PRE, CH);
    pack(nm_w2, w2n, CH, CH);
    pack(nm_w3, w3n, CH, CS);
    pack(em_w1, w1e, PRE, CH);
    pack(em_w2, w2e, CH, CH);
    pack(em_w3, w3e, CH, CZ);
    pack(ffn_w1, fw1, CS, 4 * CS);
    pack(ffn_w2, fw2, 4 * CS, CS);

    cnt_kernel<<<(N_EDGES + 255) / 256, 256, 0, stream>>>(ei, cnt);

    // ---- node path ----
    node_pts_kernel<<<N_NODES / 8, 256, 0, stream>>>(s, node_pts_w, node_pts_b,
                                                     rot, trans, ptsl, ptsg);
    edge_mlp_kernel<true><<<N_EDGES / 16, 256, 0, stream>>>(
        s, z, ei, rot, trans, ptsl, ptsg,
        w1n, nm_b1, w2n, nm_b2, w3n, nm_b3,
        nullptr, nullptr, upd, nullptr);
    node_ln1_kernel<<<N_NODES / 8, 256, 0, stream>>>(s, upd, cnt, mask,
                                                     ln1_g, ln1_b, smid);
    ffn_kernel<<<N_NODES / 16, 256, 0, stream>>>(smid, fw1, ffn_b1, fw2, ffn_b2,
                                                 ln2_g, ln2_b, out_s);

    // ---- edge path (uses updated s from d_out) ----
    node_pts_kernel<<<N_NODES / 8, 256, 0, stream>>>(out_s, edge_pts_w, edge_pts_b,
                                                     rot, trans, ptsl, ptsg);
    edge_mlp_kernel<false><<<N_EDGES / 16, 256, 0, stream>>>(
        out_s, z, ei, rot, trans, ptsl, ptsg,
        w1e, em_b1, w2e, em_b2, w3e, em_b3,
        lne_g, lne_b, nullptr, out_z);
}
